// DiGCN_IB_3MixBN_SymCat_46746424050305
// MI455X (gfx1250) — matrix operand order repaired and checked by reading, not compiled, not run on hardware
//
#include <hip/hip_runtime.h>

// ---------------------------------------------------------------------------
// DiGCN_IB_3MixBN_SymCat forward for gfx1250 (MI455X).
//
// GEMMs (~16 GFLOP total) run on v_wmma_f32_16x16x32_f16 (f16 in, f32 acc),
// fully unrolled over K (4 or 8 WMMAs straight-line per wave-tile).
// Scatter passes (~7 GB of L2-resident irregular traffic) use coalesced
// 512B-row gathers + global_atomic_add_f32 (unsafeAtomicAdd).
// ---------------------------------------------------------------------------

typedef __attribute__((ext_vector_type(16))) _Float16 v16h;
typedef __attribute__((ext_vector_type(8)))  float    v8f;

#define WPB 4   // waves per block in the GEMM kernel (128 threads, wave32)

// ---------------------------- utility kernels ------------------------------

__global__ void k_fill_zero(float* __restrict__ p, size_t n) {
  size_t i = (size_t)blockIdx.x * blockDim.x + threadIdx.x;
  if (i < n) p[i] = 0.0f;
}

__global__ void k_add3(const float* __restrict__ a, const float* __restrict__ b,
                       const float* __restrict__ c, float* __restrict__ o, int n) {
  int i = blockIdx.x * blockDim.x + threadIdx.x;
  if (i < n) o[i] = a[i] + b[i] + c[i];
}

// Convert a weight matrix to the B-operand layout: Bp[col*K + k] (f16).
// transposed==0 : W is [C,K] row-major (torch Linear, used as x @ W^T)
// transposed==1 : W is [K,C] row-major (DIGCN, used as x @ W)
__global__ void k_prep_w(const float* __restrict__ W, _Float16* __restrict__ Bp,
                         int K, int C, int transposed) {
  int idx = blockIdx.x * blockDim.x + threadIdx.x;
  if (idx >= K * C) return;
  int c = idx / K;
  int k = idx - c * K;
  float v = transposed ? W[(size_t)k * C + c] : W[(size_t)c * K + k];
  Bp[idx] = (_Float16)v;
}

// deg[src[e]] += w[e]   (w==nullptr -> 1.0)
__global__ void k_deg_accum(const int* __restrict__ src, const float* __restrict__ w,
                            float* __restrict__ deg, int nE) {
  int e = blockIdx.x * blockDim.x + threadIdx.x;
  if (e >= nE) return;
  unsafeAtomicAdd(&deg[src[e]], w ? w[e] : 1.0f);
}

// in-place deg -> dinv = deg>0 ? rsqrt(max(deg,1e-30)) : 0
__global__ void k_deg_to_dinv(float* __restrict__ p, int n) {
  int i = blockIdx.x * blockDim.x + threadIdx.x;
  if (i >= n) return;
  float d = p[i];
  p[i] = (d > 0.0f) ? rsqrtf(fmaxf(d, 1e-30f)) : 0.0f;
}

// out[dst[e], :] += X[src[e], :] * norm,  norm = (dinv? dinv[s]*w*dinv[d] : w)
// blockDim.x == F (feature dim), blockDim.y edges per block.
template <int F>
__global__ void k_scatter(const float* __restrict__ X, const int* __restrict__ eidx,
                          const float* __restrict__ w, const float* __restrict__ dinv,
                          float* __restrict__ out, int nE) {
  int e = blockIdx.x * blockDim.y + threadIdx.y;
  if (e >= nE) return;
  int s = eidx[e];
  int d = eidx[nE + e];
  float nrm = w ? w[e] : 1.0f;
  if (dinv) nrm = dinv[s] * nrm * dinv[d];
  int f = threadIdx.x;
  unsafeAtomicAdd(out + (size_t)d * F + f, X[(size_t)s * F + f] * nrm);
}

// ------------------------------ WMMA GEMM ----------------------------------
// C[M,Nc] = act( A[M,K] * B + bias ) with B pre-packed as Bp[col*K + k] f16.
// CONCAT: logical A row = [A row (KA cols) | A2 row (K-KA cols)], both with
// row stride KA (KA == K-KA == 128 in our use, K slice never straddles).
// One wave computes one 16x16 output tile; K fully unrolled (K/32 WMMAs).
template <int K, int KA, bool CONCAT, bool HAS_BIAS, bool RELU_OUT, bool RELU_A>
__global__ void __launch_bounds__(128) k_wmma_gemm(
    const float* __restrict__ A, const float* __restrict__ A2,
    const _Float16* __restrict__ B, const float* __restrict__ bias,
    float* __restrict__ C, int Mrows, int Nc) {
  const int wave = threadIdx.x >> 5;
  const int lane = threadIdx.x & 31;
  const int ntiles = Nc >> 4;
  const int tile = blockIdx.x * WPB + wave;
  const int mtile = tile / ntiles;
  const int ntile = tile - mtile * ntiles;
  if (mtile * 16 >= Mrows) return;

  const int hi  = lane >> 4;    // half-wave select
  const int l15 = lane & 15;
  int rowA = mtile * 16 + l15;  // A lane holds matrix row (l15)
  if (rowA >= Mrows) rowA = Mrows - 1;   // clamp loads on a partial tile
  const int colB = ntile * 16 + l15;     // B lane holds matrix column (l15)

  v8f acc = {0.f, 0.f, 0.f, 0.f, 0.f, 0.f, 0.f, 0.f};
  const _Float16* Bp = B + (size_t)colB * K;

#pragma unroll
  for (int kt = 0; kt < K; kt += 32) {
    // Whole 32-wide K slice lives in A (k < KA) or A2 (k >= KA): compile-time
    // per unrolled iteration since kt is constant and KA % 32 == 0.
    const float* base = (!CONCAT || (kt + 31) < KA)
                            ? (A  + (size_t)rowA * KA + kt)
                            : (A2 + (size_t)rowA * KA + (kt - KA));
    // A k order used here (B matches): lane<16 holds K {kt..kt+15},
    // lane>=16 holds K {kt+16..kt+31}  -> two 8-float chunks.
    v8f a0 = *(const v8f*)(base + 16 * hi);
    v8f a1 = *(const v8f*)(base + 16 * hi + 8);
    v16h aV;
#pragma unroll
    for (int i = 0; i < 8; ++i) {
      float f0 = a0[i], f1 = a1[i];
      if (RELU_A) { f0 = fmaxf(f0, 0.0f); f1 = fmaxf(f1, 0.0f); }
      aV[i]     = (_Float16)f0;
      aV[8 + i] = (_Float16)f1;
    }
    // B k order used here (matches A): lane<16 holds K {kt..kt+15}, lane>=16 {kt+16..kt+31}
    // at fixed column -> one contiguous 32B v16h load from the packed layout.
    const v16h bV = *(const v16h*)(Bp + kt + 16 * hi);
    acc = __builtin_amdgcn_wmma_f32_16x16x32_f16(
        /*neg_a=*/false, aV, /*neg_b=*/false, bV,
        /*c_mod=*/(short)0, acc, /*reuse_a=*/false, /*reuse_b=*/false);
  }

  const float bn = HAS_BIAS ? bias[colB] : 0.0f;
  // D layout: VGPR r -> row (r + 8*hi), lane l15 -> column.
  if (mtile * 16 + 16 <= Mrows) {           // full tile: unguarded stores
#pragma unroll
    for (int r = 0; r < 8; ++r) {
      float v = acc[r] + bn;
      if (RELU_OUT) v = fmaxf(v, 0.0f);
      C[(size_t)(mtile * 16 + r + 8 * hi) * Nc + colB] = v;
    }
  } else {
#pragma unroll
    for (int r = 0; r < 8; ++r) {
      int m = mtile * 16 + r + 8 * hi;
      if (m < Mrows) {
        float v = acc[r] + bn;
        if (RELU_OUT) v = fmaxf(v, 0.0f);
        C[(size_t)m * Nc + colB] = v;
      }
    }
  }
}

// ------------------------------ host helpers -------------------------------

static inline int gemm_blocks(int Mrows, int Nc) {
  int mtiles = (Mrows + 15) / 16;
  int tiles = mtiles * (Nc / 16);
  return (tiles + WPB - 1) / WPB;
}

static inline void scatter(const float* X, const int* eidx, const float* w,
                           const float* dinv, float* out, int E, int F,
                           hipStream_t s) {
  if (F == 128) {
    dim3 blk(128, 2);
    k_scatter<128><<<(E + 1) / 2, blk, 0, s>>>(X, eidx, w, dinv, out, E);
  } else {
    dim3 blk(64, 4);
    k_scatter<64><<<(E + 3) / 4, blk, 0, s>>>(X, eidx, w, dinv, out, E);
  }
}

static inline void fillz(float* p, size_t n, hipStream_t s) {
  int blocks = (int)((n + 255) / 256);
  k_fill_zero<<<blocks, 256, 0, s>>>(p, n);
}

static inline void prepw(const float* W, _Float16* Bp, int K, int C, int tr,
                         hipStream_t s) {
  int n = K * C;
  k_prep_w<<<(n + 255) / 256, 256, 0, s>>>(W, Bp, K, C, tr);
}

// ------------------------------ entry point --------------------------------

extern "C" void kernel_launch(void* const* d_in, const int* in_sizes, int n_in,
                              void* d_out, int out_size, void* d_ws, size_t ws_size,
                              hipStream_t stream) {
  const float* x        = (const float*)d_in[0];
  const int*   ei       = (const int*)  d_in[1];
  const int*   e_in     = (const int*)  d_in[2];
  const float* in_w     = (const float*)d_in[3];
  const int*   e_out    = (const int*)  d_in[4];
  const float* out_w    = (const float*)d_in[5];
  const int*   ei_ib    = (const int*)  d_in[6];
  const float* w_ib     = (const float*)d_in[7];
  const int*   ei2_ib   = (const int*)  d_in[8];
  const float* w2_ib    = (const float*)d_in[9];
  const float* lin1_w   = (const float*)d_in[10];
  const float* lin2_w   = (const float*)d_in[11];
  const float* ib1_ln_w = (const float*)d_in[12];
  const float* ib1_ln_b = (const float*)d_in[13];
  const float* ib1_c1_w = (const float*)d_in[14];
  const float* ib1_c1_b = (const float*)d_in[15];
  const float* ib1_c2_w = (const float*)d_in[16];
  const float* ib1_c2_b = (const float*)d_in[17];
  const float* ib2_ln_w = (const float*)d_in[18];
  const float* ib2_ln_b = (const float*)d_in[19];
  const float* ib2_c1_w = (const float*)d_in[20];
  const float* ib2_c1_b = (const float*)d_in[21];
  const float* ib2_c2_w = (const float*)d_in[22];
  const float* ib2_c2_b = (const float*)d_in[23];
  const float* conv1_w  = (const float*)d_in[24];
  const float* conv1_b  = (const float*)d_in[25];

  const int IN  = 128, HID = 128, OUT = 64;
  const int N = in_sizes[0] / IN;      // 50000
  const int E = in_sizes[3];           // 800000
  float* out = (float*)d_out;

  // ---- workspace layout (floats) ----
  float* fws = (float*)d_ws;
  const size_t NF = (size_t)N * HID;
  float* bufA = fws;                 // symx / post-conv1 h
  float* bufB = fws + 1 * NF;        // sym-branch accumulator
  float* bufC = fws + 2 * NF;        // inception accumulator
  float* bufD = fws + 3 * NF;        // xw temporary
  float* bufE = fws + 4 * NF;        // final symx [N, OUT]
  float* dinvS = fws + 4 * NF + (size_t)N * OUT;   // 4 contiguous [N] arrays
  float* dinvI = dinvS + N;
  float* dinvO = dinvI + N;
  float* dinvB = dinvO + N;
  float* bias1 = dinvB + N;          // folded IB1 bias (128)
  float* bias2 = bias1 + HID;        // folded IB2 bias (128)
  size_t hwoff = (size_t)(bias2 + HID - fws);
  hwoff = (hwoff + 15) & ~(size_t)15;              // 64B-align the f16 region
  _Float16* hw       = (_Float16*)(fws + hwoff);
  _Float16* hw_lin1  = hw;                          // 128*128
  _Float16* hw_ib1ln = hw_lin1  + HID * IN;
  _Float16* hw_ib1c1 = hw_ib1ln + HID * IN;
  _Float16* hw_ib1c2 = hw_ib1c1 + HID * IN;
  _Float16* hw_ib2ln = hw_ib1c2 + HID * IN;
  _Float16* hw_ib2c1 = hw_ib2ln + HID * HID;
  _Float16* hw_ib2c2 = hw_ib2c1 + HID * HID;
  _Float16* hw_conv1 = hw_ib2c2 + HID * HID;        // 128*256
  _Float16* hw_lin2  = hw_conv1 + HID * 2 * HID;    // 64*128

  // ---- zero accumulators + degree arrays + output ----
  fillz(bufB, NF, stream);
  fillz(out, (size_t)N * OUT, stream);
  fillz(dinvS, (size_t)4 * N, stream);

  // ---- pack weights to f16 B-operand layout ----
  prepw(lin1_w,   hw_lin1,  IN,      HID, 0, stream);  // x @ W^T
  prepw(ib1_ln_w, hw_ib1ln, IN,      HID, 0, stream);
  prepw(ib1_c1_w, hw_ib1c1, IN,      HID, 1, stream);  // x @ W
  prepw(ib1_c2_w, hw_ib1c2, IN,      HID, 1, stream);
  prepw(ib2_ln_w, hw_ib2ln, HID,     HID, 0, stream);
  prepw(ib2_c1_w, hw_ib2c1, HID,     HID, 1, stream);
  prepw(ib2_c2_w, hw_ib2c2, HID,     HID, 1, stream);
  prepw(conv1_w,  hw_conv1, 2 * HID, HID, 0, stream);
  prepw(lin2_w,   hw_lin2,  HID,     OUT, 0, stream);

  // ---- fold the three per-block biases into one vector each ----
  k_add3<<<1, HID, 0, stream>>>(ib1_ln_b, ib1_c1_b, ib1_c2_b, bias1, HID);
  k_add3<<<1, HID, 0, stream>>>(ib2_ln_b, ib2_c1_b, ib2_c2_b, bias2, HID);

  // ---- degrees (by src) -> dinv for the 4 normalized edge sets ----
  int eb = (E + 255) / 256;
  k_deg_accum<<<eb, 256, 0, stream>>>(ei,    nullptr, dinvS, E);
  k_deg_accum<<<eb, 256, 0, stream>>>(e_in,  in_w,    dinvI, E);
  k_deg_accum<<<eb, 256, 0, stream>>>(e_out, out_w,   dinvO, E);
  k_deg_accum<<<eb, 256, 0, stream>>>(ei_ib, nullptr, dinvB, E);
  k_deg_to_dinv<<<(4 * N + 255) / 256, 256, 0, stream>>>(dinvS, 4 * N);

  const int g128 = gemm_blocks(N, HID);
  const int g64  = gemm_blocks(N, OUT);

  // ---- stage A/B: symx = x @ lin1^T ; sym = sum of 3 DGCN scatters ----
  k_wmma_gemm<128, 128, false, false, false, false><<<g128, 128, 0, stream>>>(
      x, nullptr, hw_lin1, nullptr, bufA, N, HID);
  scatter(bufA, ei,    nullptr, dinvS, bufB, E, HID, stream);
  scatter(bufA, e_in,  in_w,    dinvI, bufB, E, HID, stream);
  scatter(bufA, e_out, out_w,   dinvO, bufB, E, HID, stream);

  // ---- stage C: InceptionBlock 1 -> bufC ----
  k_wmma_gemm<128, 128, false, true, false, false><<<g128, 128, 0, stream>>>(
      x, nullptr, hw_ib1ln, bias1, bufC, N, HID);
  k_wmma_gemm<128, 128, false, false, false, false><<<g128, 128, 0, stream>>>(
      x, nullptr, hw_ib1c1, nullptr, bufD, N, HID);
  scatter(bufD, ei_ib,  w_ib,  nullptr, bufC, E, HID, stream);
  k_wmma_gemm<128, 128, false, false, false, false><<<g128, 128, 0, stream>>>(
      x, nullptr, hw_ib1c2, nullptr, bufD, N, HID);
  scatter(bufD, ei2_ib, w2_ib, nullptr, bufC, E, HID, stream);

  // ---- stage D: h = relu([bufC | bufB] @ conv1^T + b) -> bufA ----
  k_wmma_gemm<256, 128, true, true, true, false><<<g128, 128, 0, stream>>>(
      bufC, bufB, hw_conv1, conv1_b, bufA, N, HID);

  // ---- stage E: InceptionBlock 2 on bufA -> bufC (pre-ReLU sum) ----
  k_wmma_gemm<128, 128, false, true, false, false><<<g128, 128, 0, stream>>>(
      bufA, nullptr, hw_ib2ln, bias2, bufC, N, HID);
  k_wmma_gemm<128, 128, false, false, false, false><<<g128, 128, 0, stream>>>(
      bufA, nullptr, hw_ib2c1, nullptr, bufD, N, HID);
  scatter(bufD, ei_ib,  w_ib,  nullptr, bufC, E, HID, stream);
  k_wmma_gemm<128, 128, false, false, false, false><<<g128, 128, 0, stream>>>(
      bufA, nullptr, hw_ib2c2, nullptr, bufD, N, HID);
  scatter(bufD, ei2_ib, w2_ib, nullptr, bufC, E, HID, stream);

  // ---- stage F: symx2 = relu(bufC) @ lin2^T -> bufE (ReLU fused on A) ----
  k_wmma_gemm<128, 128, false, false, false, true><<<g64, 128, 0, stream>>>(
      bufC, nullptr, hw_lin2, nullptr, bufE, N, OUT);

  // ---- stage G: out = 3 DGCN scatters of bufE (note edge_index_ib here) ----
  scatter(bufE, ei_ib, nullptr, dinvB, out, E, OUT, stream);
  scatter(bufE, e_in,  in_w,    dinvI, out, E, OUT, stream);
  scatter(bufE, e_out, out_w,   dinvO, out, E, OUT, stream);
}
